// CausalSelfAttention_14774687498766
// MI455X (gfx1250) — matrix operand order repaired and checked by reading, not compiled, not run on hardware
//
#include <hip/hip_runtime.h>

// ---------------- problem constants ----------------
constexpr int BDIM = 2;          // batch
constexpr int T    = 2048;       // seq len
constexpr int DIM  = 2048;
constexpr int NH   = 16;
constexpr int HD   = 128;        // head dim
constexpr int M    = BDIM * T;   // 4096 rows
constexpr float RESTORE_SCALE = 45.25483399593904f;   // sqrt(2048)
constexpr float ATTN_SCALE    = 11.313708498984761f;  // sqrt(128)
constexpr float LN10000_OVER_64 = 0.14391156809932845f;

typedef __bf16 bf16;
typedef __attribute__((ext_vector_type(16))) __bf16 v16bf;
typedef __attribute__((ext_vector_type(8)))  __bf16 v8bf;
typedef __attribute__((ext_vector_type(8)))  float  v8f;

#define WMMA_BF16(a, b, c) \
  __builtin_amdgcn_wmma_f32_16x16x32_bf16(false, (a), false, (b), (short)0, (c), false, false)

// ---------------- CDNA5 data-movement helpers ----------------
#if __has_builtin(__builtin_amdgcn_tensor_load_to_lds)
#define HAVE_TDM 1
#else
#define HAVE_TDM 0
#endif

#if __has_builtin(__builtin_amdgcn_s_wait_tensorcnt)
#define WAIT_TENSORCNT(n) __builtin_amdgcn_s_wait_tensorcnt((short)(n))
#else
#define WAIT_TENSORCNT(n) asm volatile("s_wait_tensorcnt %0" :: "i"(n) : "memory")
#endif

#define WAIT_ASYNCCNT0() asm volatile("s_wait_asynccnt 0" ::: "memory")

__device__ __forceinline__ unsigned lds_off(const void* p) {
  return (unsigned)(size_t)p;   // generic ptr low 32 bits == wave-relative LDS byte offset
}

// per-lane async 16B copy: global -> LDS (ASYNCcnt tracked)
__device__ __forceinline__ void async_copy16(const void* gsrc, void* ldst) {
  unsigned l = lds_off(ldst);
  asm volatile("global_load_async_to_lds_b128 %0, %1, off"
               :: "v"(l), "v"(gsrc) : "memory");
}

#if HAVE_TDM
typedef unsigned int u32x4 __attribute__((ext_vector_type(4)));
typedef int          i32x4 __attribute__((ext_vector_type(4)));
typedef int          i32x8 __attribute__((ext_vector_type(8)));

// TDM: DMA a 2D tile (tile_d0 elems/row x tile_d1 rows, 2-byte elements) from a
// row-major tensor (tensor_d0 x tensor_d1, row stride stride_d0 elems) into LDS,
// inserting pad_amount(code) DWORDs every pad_interval(code) DWORDs.
__device__ __forceinline__ void tdm_load_tile_2d(
    const void* gsrc, unsigned ldsoff,
    unsigned tensor_d0, unsigned tensor_d1, unsigned stride_d0,
    unsigned tile_d0, unsigned tile_d1,
    unsigned pad_interval_code, unsigned pad_amount_code) {
  unsigned long long ga = (unsigned long long)(size_t)gsrc;
  u32x4 g0;
  g0[0] = 1u;                                            // count=1, user descriptor
  g0[1] = ldsoff;                                        // lds_addr
  g0[2] = (unsigned)ga;                                  // global_addr[95:64]
  g0[3] = (unsigned)((ga >> 32) & 0x01FFFFFFu) | 0x80000000u;   // addr msbs | type=2
  i32x8 g1;
  g1[0] = (int)((1u << 16) |                             // data_size = 2 bytes
                (1u << 20) |                             // pad_enable
                (pad_interval_code << 22) |
                (pad_amount_code << 25));
  g1[1] = (int)((tensor_d0 & 0xFFFFu) << 16);            // tensor_dim0[15:0]
  g1[2] = (int)((tensor_d0 >> 16) | ((tensor_d1 & 0xFFFFu) << 16));
  g1[3] = (int)((tensor_d1 >> 16) | ((tile_d0 & 0xFFFFu) << 16));
  g1[4] = (int)(tile_d1 & 0xFFFFu);                      // tile_dim1 (tile_dim2=0)
  g1[5] = (int)stride_d0;                                // tensor_dim0_stride lo32
  g1[6] = 0;
  g1[7] = 0;
  i32x4 z4 = {};
#if __clang_major__ >= 23
  i32x8 z8 = {};
  __builtin_amdgcn_tensor_load_to_lds(g0, g1, z4, z4, z8, 0);
#else
  __builtin_amdgcn_tensor_load_to_lds(g0, g1, z4, z4, 0);
#endif
}
#endif  // HAVE_TDM

__device__ __forceinline__ bf16 to_bf16(float f) {
  unsigned u = __builtin_bit_cast(unsigned, f);
  unsigned r = u + 0x7FFFu + ((u >> 16) & 1u);   // round-to-nearest-even
  unsigned short h = (unsigned short)(r >> 16);
  return __builtin_bit_cast(bf16, h);
}

// A fragment (16x32, M x K): lane holds row m = lane&15;
// k in [kb, kb+8) U [16+kb, 16+kb+8), kb = 8*(lane>>4).  tile is [row][k], stride ld.
__device__ __forceinline__ v16bf frag_A(const bf16* tile, int ld, int lane) {
  const int m  = lane & 15;
  const int kb = (lane >> 4) << 3;
  const bf16* p = tile + m * ld + kb;
  v8bf lo = *(const v8bf*)(p);
  v8bf hi = *(const v8bf*)(p + 16);
  v16bf a;
#pragma unroll
  for (int j = 0; j < 8; ++j) { a[j] = lo[j]; a[j + 8] = hi[j]; }
  return a;
}

// B fragment (32x16, K x N): lane holds column n = lane&15;
// k in [kb, kb+8) U [16+kb, 16+kb+8), kb = 8*(lane>>4).  tile stores column n as row n (k-major), stride ld.
__device__ __forceinline__ v16bf frag_B(const bf16* tile, int ld, int lane) {
  const int n  = lane & 15;
  const int kb = (lane >> 4) << 3;
  const bf16* p = tile + n * ld + kb;
  v8bf lo = *(const v8bf*)(p);
  v8bf hi = *(const v8bf*)(p + 16);
  v16bf b;
#pragma unroll
  for (int j = 0; j < 8; ++j) { b[j] = lo[j]; b[j + 8] = hi[j]; }
  return b;
}

// ---------------- f32 -> bf16 convert ----------------
__global__ void cvt_bf16_kernel(const float* __restrict__ s, bf16* __restrict__ d, int n) {
  int i = blockIdx.x * blockDim.x + threadIdx.x;
  int stride = gridDim.x * blockDim.x;
  for (; i < n; i += stride) d[i] = to_bf16(s[i]);
}

// ---------------- bf16 GEMM: C = A(MxK) @ W(NxK)^T ----------------
// mode 0: Q (RoPE+sqk, bf16 out [B][H][T][D]);  mode 1: K (same)
// mode 2: V (plain bf16 out [B][H][T][D]);      mode 3: f32 out row-major (M x DIM)
#define G_LDT 40                     // 32 + 8 pad (bf16 elems) == 64B data + 16B pad
#define G_BUFB 20480                 // bytes per double-buffer slab (Xs 10240 + Ws 10240)
#define G_SMEM_BYTES (128 * 132 * 4) // f32 epilogue tile; aliases the staging buffers

__global__ __launch_bounds__(256) void gemm_bf16_kernel(
    const bf16* __restrict__ A, const bf16* __restrict__ W,
    void* __restrict__ out, const float* __restrict__ sqk, int mode) {
  extern __shared__ char smem[];
  float* Cs = (float*)smem;            // 128 x 132, used after the k-loop

  const int tid = threadIdx.x;
  const int lane = tid & 31, w = tid >> 5;
  const int wm = w >> 2, wn = w & 3;               // 2 x 4 wave grid; wave tile 64 x 32
  const int m0 = blockIdx.y * 128, n0 = blockIdx.x * 128;

  v8f acc[4][2];
#pragma unroll
  for (int ms = 0; ms < 4; ++ms)
#pragma unroll
    for (int ns = 0; ns < 2; ++ns) acc[ms][ns] = (v8f){};

#if HAVE_TDM
  // ---- TDM double-buffered k-loop: wave 0 drives the Tensor Data Mover ----
  const unsigned sb = lds_off(smem);
  if (w == 0) {
    tdm_load_tile_2d(A + (size_t)m0 * DIM, sb,         DIM, M,   DIM, 32, 128, 3, 3);
    tdm_load_tile_2d(W + (size_t)n0 * DIM, sb + 10240, DIM, DIM, DIM, 32, 128, 3, 3);
  }
  for (int k0 = 0; k0 < DIM; k0 += 32) {
    const int cur = (k0 >> 5) & 1;
    if (w == 0) {
      if (k0 + 32 < DIM) {   // prefetch next slab into the other buffer
        const unsigned nb = sb + (cur ? 0u : (unsigned)G_BUFB);
        tdm_load_tile_2d(A + (size_t)m0 * DIM + k0 + 32, nb,
                         DIM, M, DIM, 32, 128, 3, 3);
        tdm_load_tile_2d(W + (size_t)n0 * DIM + k0 + 32, nb + 10240,
                         DIM, DIM, DIM, 32, 128, 3, 3);
        WAIT_TENSORCNT(2);   // current slab complete, next still in flight
      } else {
        WAIT_TENSORCNT(0);
      }
    }
    __syncthreads();
    const bf16* Xs = (const bf16*)(smem + (cur ? G_BUFB : 0));
    const bf16* Ws = (const bf16*)(smem + (cur ? G_BUFB : 0) + 10240);
    v16bf b0 = frag_B(Ws + (wn * 32) * G_LDT, G_LDT, lane);
    v16bf b1 = frag_B(Ws + (wn * 32 + 16) * G_LDT, G_LDT, lane);
#pragma unroll
    for (int ms = 0; ms < 4; ++ms) {
      v16bf a = frag_A(Xs + (wm * 64 + ms * 16) * G_LDT, G_LDT, lane);
      acc[ms][0] = WMMA_BF16(a, b0, acc[ms][0]);
      acc[ms][1] = WMMA_BF16(a, b1, acc[ms][1]);
    }
    __syncthreads();
  }
#else
  // ---- fallback: cooperative VGPR staging ----
  for (int k0 = 0; k0 < DIM; k0 += 32) {
    bf16* Xs = (bf16*)smem;
    bf16* Ws = (bf16*)(smem + 10240);
#pragma unroll
    for (int t = tid; t < 512; t += 256) {
      int r = t >> 2, c = (t & 3) << 3;
      *(v8bf*)&Xs[r * G_LDT + c] = *(const v8bf*)&A[(size_t)(m0 + r) * DIM + k0 + c];
      *(v8bf*)&Ws[r * G_LDT + c] = *(const v8bf*)&W[(size_t)(n0 + r) * DIM + k0 + c];
    }
    __syncthreads();
    v16bf b0 = frag_B(Ws + (wn * 32) * G_LDT, G_LDT, lane);
    v16bf b1 = frag_B(Ws + (wn * 32 + 16) * G_LDT, G_LDT, lane);
#pragma unroll
    for (int ms = 0; ms < 4; ++ms) {
      v16bf a = frag_A(Xs + (wm * 64 + ms * 16) * G_LDT, G_LDT, lane);
      acc[ms][0] = WMMA_BF16(a, b0, acc[ms][0]);
      acc[ms][1] = WMMA_BF16(a, b1, acc[ms][1]);
    }
    __syncthreads();
  }
#endif

  // dump accumulators (C layout: n = lane&15, m = i + 8*(lane>>4)) into f32 LDS tile
#pragma unroll
  for (int ms = 0; ms < 4; ++ms)
#pragma unroll
    for (int ns = 0; ns < 2; ++ns)
#pragma unroll
      for (int i = 0; i < 8; ++i) {
        int r = wm * 64 + ms * 16 + i + 8 * (lane >> 4);
        int c = wn * 32 + ns * 16 + (lane & 15);
        Cs[r * 132 + c] = acc[ms][ns][i];
      }
  __syncthreads();

  if (mode == 3) {
    float* y = (float*)out;
    for (int e = tid; e < 128 * 128; e += 256) {
      int r = e >> 7, c = e & 127;
      y[(size_t)(m0 + r) * DIM + n0 + c] = Cs[r * 132 + c];
    }
  } else {
    bf16* o = (bf16*)out;
    const int h = blockIdx.x;                     // N-tile == one head
    for (int e = tid; e < 128 * 64; e += 256) {
      int r = e >> 6, j = e & 63;
      float xr = Cs[r * 132 + j];
      float xi = Cs[r * 132 + j + 64];
      int mg = m0 + r, bb = mg >> 11, tp = mg & (T - 1);
      size_t base = (((size_t)bb * NH + h) * T + tp) * HD;
      if (mode == 2) {
        o[base + j]      = to_bf16(xr);
        o[base + j + 64] = to_bf16(xi);
      } else {
        float th = __expf((float)j * -LN10000_OVER_64);   // 10000^(-j/64)
        float ang = (float)tp * th, s_, c_;
        __sincosf(ang, &s_, &c_);
        float sr = sqk[h * HD + j]      * RESTORE_SCALE;
        float si = sqk[h * HD + j + 64] * RESTORE_SCALE;
        o[base + j]      = to_bf16((xr * c_ - xi * s_) * sr);
        o[base + j + 64] = to_bf16((xr * s_ + xi * c_) * si);
      }
    }
  }
}

// ---------------- flash attention (bf16, online softmax) ----------------
// grid: (T/128, B*NH). 256 threads = 8 waves; wave w owns 16 query rows.
#define F_SMEM_BYTES ((128 * 136 + 32 * 136 + 128 * 40 + 8 * 16 * 40) * 2)

__global__ __launch_bounds__(256) void flash_attn_kernel(
    const bf16* __restrict__ Q, const bf16* __restrict__ K,
    const bf16* __restrict__ V, bf16* __restrict__ O) {
  extern __shared__ char smem[];
  bf16* Qs  = (bf16*)smem;            // 128 x 136
  bf16* Ks  = Qs + 128 * 136;         // 32 x 136
  bf16* Vst = Ks + 32 * 136;          // 128 x 40   (V transposed: [d][s])
  bf16* Pb  = Vst + 128 * 40;         // 8 waves x 16 x 40

  const int tid = threadIdx.x, lane = tid & 31, w = tid >> 5;
  const int hi = lane >> 4, ln = lane & 15;
  const int m0 = blockIdx.x * 128;
  const int bh = blockIdx.y;
  const size_t sbase = (size_t)bh * T * HD;

  // async global->LDS staging of the query tile
  for (int t = tid; t < 128 * 16; t += 256) {
    int r = t >> 4, c = (t & 15) << 3;
    async_copy16(&Q[sbase + (size_t)(m0 + r) * HD + c], &Qs[r * 136 + c]);
  }
  WAIT_ASYNCCNT0();

  v8f acc[8];
#pragma unroll
  for (int dn = 0; dn < 8; ++dn) acc[dn] = (v8f){};
  float rmax[8], rsum[8];
#pragma unroll
  for (int i = 0; i < 8; ++i) { rmax[i] = -3.0e38f; rsum[i] = 0.0f; }
  const int qr0 = m0 + w * 16;

  for (int j0 = 0; j0 < m0 + 128; j0 += 32) {
    __syncthreads();   // previous chunk fully consumed before overwriting Ks/Vst
#pragma unroll
    for (int t = tid; t < 512; t += 256) {
      int r = t >> 4, c = (t & 15) << 3;
      async_copy16(&K[sbase + (size_t)(j0 + r) * HD + c], &Ks[r * 136 + c]);
      v8bf vv = *(const v8bf*)&V[sbase + (size_t)(j0 + r) * HD + c];
#pragma unroll
      for (int u = 0; u < 8; ++u) Vst[(c + u) * 40 + r] = vv[u];
    }
    WAIT_ASYNCCNT0();
    __syncthreads();

    if (j0 <= qr0 + 15) {              // chunk intersects this wave's causal region
      v8f S0 = (v8f){}, S1 = (v8f){};
#pragma unroll
      for (int dc = 0; dc < 128; dc += 32) {
        v16bf a  = frag_A(Qs + (w * 16) * 136 + dc, 136, lane);
        v16bf b0 = frag_B(Ks + dc, 136, lane);
        v16bf b1 = frag_B(Ks + 16 * 136 + dc, 136, lane);
        S0 = WMMA_BF16(a, b0, S0);
        S1 = WMMA_BF16(a, b1, S1);
      }
      // scale + causal mask + per-chunk row max
      float cm[8];
#pragma unroll
      for (int i = 0; i < 8; ++i) {
        int mr = qr0 + i + 8 * hi;
        float v0 = S0[i] * ATTN_SCALE;
        float v1 = S1[i] * ATTN_SCALE;
        v0 = (j0 + ln      <= mr) ? v0 : -3.0e38f;
        v1 = (j0 + 16 + ln <= mr) ? v1 : -3.0e38f;
        S0[i] = v0; S1[i] = v1;
        cm[i] = fmaxf(v0, v1);
      }
#pragma unroll
      for (int o = 1; o < 16; o <<= 1)
#pragma unroll
        for (int i = 0; i < 8; ++i) cm[i] = fmaxf(cm[i], __shfl_xor(cm[i], o, 32));

      float pr[8];
#pragma unroll
      for (int i = 0; i < 8; ++i) {
        float nm = fmaxf(rmax[i], cm[i]);
        float al = __expf(rmax[i] - nm);
        rmax[i] = nm;
        S0[i] = __expf(S0[i] - nm);
        S1[i] = __expf(S1[i] - nm);
        pr[i] = S0[i] + S1[i];
        rsum[i] *= al;
#pragma unroll
        for (int dn = 0; dn < 8; ++dn) acc[dn][i] *= al;
      }
#pragma unroll
      for (int o = 1; o < 16; o <<= 1)
#pragma unroll
        for (int i = 0; i < 8; ++i) pr[i] += __shfl_xor(pr[i], o, 32);
#pragma unroll
      for (int i = 0; i < 8; ++i) rsum[i] += pr[i];

      // transpose P (C layout -> A layout) through per-wave LDS scratch
      bf16* Pw = Pb + w * 16 * 40;
#pragma unroll
      for (int i = 0; i < 8; ++i) {
        Pw[(i + 8 * hi) * 40 + ln]      = to_bf16(S0[i]);
        Pw[(i + 8 * hi) * 40 + 16 + ln] = to_bf16(S1[i]);
      }
      asm volatile("s_wait_dscnt 0" ::: "memory");   // same-wave LDS RAW fence
      v16bf pa = frag_A(Pw, 40, lane);
#pragma unroll
      for (int dn = 0; dn < 8; ++dn) {
        v16bf vb = frag_B(Vst + dn * 16 * 40, 40, lane);
        acc[dn] = WMMA_BF16(pa, vb, acc[dn]);
      }
    }
  }

  // epilogue: divide by row sums, store O row-major (B*T, DIM) for the Wo GEMM
  const int b_ = bh >> 4, h_ = bh & 15;
#pragma unroll
  for (int dn = 0; dn < 8; ++dn)
#pragma unroll
    for (int i = 0; i < 8; ++i) {
      int tq = qr0 + i + 8 * hi;
      float o = acc[dn][i] / rsum[i];
      O[((size_t)(b_ * T + tq)) * DIM + h_ * HD + dn * 16 + ln] = to_bf16(o);
    }
}

// ---------------- in-place row L2-normalization of d_out ----------------
__global__ __launch_bounds__(256) void norm_rows_kernel(float* __restrict__ y) {
  __shared__ float red[8];
  const int tid = threadIdx.x, lane = tid & 31, w = tid >> 5;
  float* p = y + (size_t)blockIdx.x * DIM;
  float s = 0.0f;
  for (int c = tid; c < DIM; c += 256) { float v = p[c]; s += v * v; }
#pragma unroll
  for (int o = 16; o > 0; o >>= 1) s += __shfl_xor(s, o, 32);
  if (lane == 0) red[w] = s;
  __syncthreads();
  if (tid == 0) {
    float t = 0.0f;
#pragma unroll
    for (int j = 0; j < 8; ++j) t += red[j];
    red[0] = 1.0f / fmaxf(sqrtf(t), 1e-12f);
  }
  __syncthreads();
  float inv = red[0];
  for (int c = tid; c < DIM; c += 256) p[c] *= inv;
}

// ---------------- launch ----------------
extern "C" void kernel_launch(void* const* d_in, const int* in_sizes, int n_in,
                              void* d_out, int out_size, void* d_ws, size_t ws_size,
                              hipStream_t stream) {
  const float* x   = (const float*)d_in[0];
  const float* Wq  = (const float*)d_in[1];
  const float* Wk  = (const float*)d_in[2];
  const float* Wv  = (const float*)d_in[3];
  const float* Wo  = (const float*)d_in[4];
  const float* sqk = (const float*)d_in[5];

  char* ws = (char*)d_ws;
  size_t off = 0;
  auto carve = [&](size_t bytes) -> char* {
    char* p = ws + off;
    off = (off + bytes + 255) & ~(size_t)255;
    return p;
  };
  bf16* Xb  = (bf16*)carve((size_t)M * DIM * 2);
  bf16* Wqb = (bf16*)carve((size_t)DIM * DIM * 2);
  bf16* Wkb = (bf16*)carve((size_t)DIM * DIM * 2);
  bf16* Wvb = (bf16*)carve((size_t)DIM * DIM * 2);
  bf16* Wob = (bf16*)carve((size_t)DIM * DIM * 2);
  bf16* Qb  = (bf16*)carve((size_t)M * DIM * 2);
  bf16* Kb  = (bf16*)carve((size_t)M * DIM * 2);
  bf16* Vb  = (bf16*)carve((size_t)M * DIM * 2);
  bf16* Ob  = (bf16*)carve((size_t)M * DIM * 2);

  cvt_bf16_kernel<<<2048, 256, 0, stream>>>(x,  Xb,  M * DIM);
  cvt_bf16_kernel<<<1024, 256, 0, stream>>>(Wq, Wqb, DIM * DIM);
  cvt_bf16_kernel<<<1024, 256, 0, stream>>>(Wk, Wkb, DIM * DIM);
  cvt_bf16_kernel<<<1024, 256, 0, stream>>>(Wv, Wvb, DIM * DIM);
  cvt_bf16_kernel<<<1024, 256, 0, stream>>>(Wo, Wob, DIM * DIM);

  dim3 gg(DIM / 128, M / 128);   // (16, 32)
  gemm_bf16_kernel<<<gg, 256, G_SMEM_BYTES, stream>>>(Xb, Wqb, Qb, sqk, 0);
  gemm_bf16_kernel<<<gg, 256, G_SMEM_BYTES, stream>>>(Xb, Wkb, Kb, sqk, 1);
  gemm_bf16_kernel<<<gg, 256, G_SMEM_BYTES, stream>>>(Xb, Wvb, Vb, sqk, 2);

  dim3 ga(T / 128, BDIM * NH);   // (16, 32)
  flash_attn_kernel<<<ga, 256, F_SMEM_BYTES, stream>>>(Qb, Kb, Vb, Ob);

  gemm_bf16_kernel<<<gg, 256, G_SMEM_BYTES, stream>>>(Ob, Wob, d_out, sqk, 3);

  norm_rows_kernel<<<M, 256, 0, stream>>>((float*)d_out);
}
